// CWFA_11682311045887
// MI455X (gfx1250) — matrix operand order repaired and checked by reading, not compiled, not run on hardware
//
#include <hip/hip_runtime.h>

// ---------------------------------------------------------------------------
// CWFA recurrence on MI455X (gfx1250, wave32, WMMA).
//
//   M_{n,t}[i,k] = sum_j A[i,j,k] * x[n,t,j]      (parallel: 275 GFLOP GEMM)
//   h <- h^T M_{n,t}  sequentially over t         (bandwidth bound scan)
//   out = h @ Omega
//
// Phase 1 runs as bf16 WMMA (v_wmma_f32_16x16x32_bf16, fp32 accumulate),
// chunked over time so the M buffer (T_c * 4 MB bf16) stays L2-resident.
// Phase 2 streams M once (1 FMA per element). h is fp32 end-to-end.
// ---------------------------------------------------------------------------

typedef __attribute__((ext_vector_type(16))) __bf16 v16bf;
typedef __attribute__((ext_vector_type(8)))  __bf16 v8bf;
typedef __attribute__((ext_vector_type(8)))  float  v8f;
typedef unsigned short u16b;
typedef unsigned int   u32b;

__device__ __forceinline__ u16b f2bf(float f) {          // fp32 -> bf16 RNE
  u32b u = __builtin_bit_cast(u32b, f);
  u32b r = 0x7FFFu + ((u >> 16) & 1u);
  return (u16b)((u + r) >> 16);
}
__device__ __forceinline__ float bf2f(u16b h) {
  u32b u = ((u32b)h) << 16;
  return __builtin_bit_cast(float, u);
}

constexpr int Ndim = 128;   // batch
constexpr int Ldim = 512;   // sequence length
constexpr int Ddim = 128;   // input dim
constexpr int Rdim = 128;   // rank
constexpr int Mdim = 32;    // output dim
constexpr int RR   = Rdim * Rdim;   // 16384
constexpr int LDSP = 136;   // padded LDS row stride (elements) -> conflict-free

// --- A[i][j][k] (fp32) -> wsA[(i*R+k)][j] (bf16): B-operand, K(j)-contiguous
__global__ __launch_bounds__(256) void cwfa_convA(const float* __restrict__ A,
                                                  u16b* __restrict__ wsA) {
  int idx = blockIdx.x * 256 + threadIdx.x;       // 2M elements
  int j = idx & 127;
  int k = (idx >> 7) & 127;
  int i = idx >> 14;
  wsA[idx] = f2bf(A[(size_t)i * 16384 + (size_t)j * 128 + k]);
}

// --- h[n][i] = alpha[i]
__global__ __launch_bounds__(256) void cwfa_init_h(const float* __restrict__ alpha,
                                                   float* __restrict__ h) {
  int idx = blockIdx.x * 256 + threadIdx.x;       // 16384
  h[idx] = alpha[idx & 127];
}

// --- Phase 1: M[(tt*N+n)][i*R+k] = sum_j x[n,t0+tt,j] * A[i,j,k]
// Grid: (colTiles=128, rowTiles=Tc). Block: 256 threads = 8 waves.
// Each block: 128x128 output tile, K=128 in 4 WMMA k-steps.
__global__ __launch_bounds__(256) void cwfa_phase1(const float* __restrict__ x,
                                                   const u16b* __restrict__ wsA,
                                                   u16b* __restrict__ wsM,
                                                   int t0) {
  __shared__ u16b ldsX[128 * LDSP];   // X tile   [row(n)][j]       bf16
  __shared__ u16b ldsB[128 * LDSP];   // A_r tile [col(i*R+k)][j]   bf16 (B^T)
  const int tid     = threadIdx.x;
  const int colBase = blockIdx.x * 128;
  const int tt      = blockIdx.y;           // time step within chunk
  const int t       = t0 + tt;
  const int rowBase = tt * 128;             // chunk-row base (= tt*N)

  // Stage X: rows are samples n=0..127 at fixed t. fp32 -> bf16.
  #pragma unroll
  for (int it = 0; it < 16; ++it) {         // 4096 float4 / 256 threads
    int q  = tid + it * 256;
    int rl = q >> 5;                        // sample n
    int j4 = (q & 31) * 4;
    float4 v = *(const float4*)(x + ((size_t)rl * Ldim + t) * Ddim + j4);
    uint2 p;
    p.x = (u32b)f2bf(v.x) | ((u32b)f2bf(v.y) << 16);
    p.y = (u32b)f2bf(v.z) | ((u32b)f2bf(v.w) << 16);
    *(uint2*)&ldsX[rl * LDSP + j4] = p;
  }
  // Stage B^T: 128 columns of A_r, each 128 bf16 (already bf16 in ws).
  #pragma unroll
  for (int it = 0; it < 8; ++it) {          // 2048 x 16B / 256 threads
    int q  = tid + it * 256;
    int c  = q >> 4;
    int k8 = (q & 15) * 8;
    *(uint4*)&ldsB[c * LDSP + k8] =
        *(const uint4*)&wsA[(size_t)(colBase + c) * 128 + k8];
  }
  __syncthreads();

  const int  lane   = tid & 31;
  const int  wave   = tid >> 5;             // wave w -> rows [16w,16w+16)
  const int  l15    = lane & 15;
  const bool hiHalf = (lane & 16) != 0;

  v8f acc[8];
  #pragma unroll
  for (int c = 0; c < 8; ++c) acc[c] = (v8f){0.f,0.f,0.f,0.f,0.f,0.f,0.f,0.f};

  #pragma unroll
  for (int kk = 0; kk < 4; ++kk) {
    // A fragment (16x32 bf16): lanes 0-15 row m hold K {0..7,16..23},
    // lanes 16-31 hold K {8..15,24..31}  (ISA 7.12.2 16-bit A layout).
    int arow = wave * 16 + l15;
    int ak0  = kk * 32 + (hiHalf ? 8 : 0);
    v8bf alo = *(const v8bf*)&ldsX[arow * LDSP + ak0];
    v8bf ahi = *(const v8bf*)&ldsX[arow * LDSP + ak0 + 16];
    v16bf afrag = __builtin_shufflevector(alo, ahi,
        0,1,2,3,4,5,6,7,8,9,10,11,12,13,14,15);
    int bk0 = kk * 32 + (hiHalf ? 8 : 0);
    #pragma unroll
    for (int c = 0; c < 8; ++c) {
      // B fragment (32x16 bf16): lane holds column N=l15, same K order as A;
      // lanes 0-15: K {0..7,16..23}, lanes 16-31: K {8..15,24..31}.
      int bcol = c * 16 + l15;
      v8bf blo = *(const v8bf*)&ldsB[bcol * LDSP + bk0];
      v8bf bhi = *(const v8bf*)&ldsB[bcol * LDSP + bk0 + 16];
      v16bf bfrag = __builtin_shufflevector(blo, bhi,
          0,1,2,3,4,5,6,7,8,9,10,11,12,13,14,15);
      acc[c] = __builtin_amdgcn_wmma_f32_16x16x32_bf16(
          false, afrag, false, bfrag, (short)0, acc[c], false, false);
    }
  }
  __syncthreads();                          // all waves done reading ldsX

  // D fragments -> LDS (bf16), ISA C/D layout: VGPR v = row v (+8 hi half).
  #pragma unroll
  for (int c = 0; c < 8; ++c) {
    int col = c * 16 + l15;
    #pragma unroll
    for (int v = 0; v < 8; ++v) {
      int row = wave * 16 + v + (hiHalf ? 8 : 0);
      ldsX[row * LDSP + col] = f2bf(acc[c][v]);
    }
  }
  __syncthreads();

  // Coalesced 16B stores of the 128x128 bf16 tile into wsM.
  #pragma unroll
  for (int it = 0; it < 8; ++it) {
    int q  = tid + it * 256;
    int rl = q >> 4;
    int c8 = (q & 15) * 8;
    *(uint4*)(wsM + (size_t)(rowBase + rl) * RR + colBase + c8) =
        *(const uint4*)&ldsX[rl * LDSP + c8];
  }
}

// --- Phase 2: sequential scan within a chunk. One block per sample n.
// h'[k] = sum_i h[i] * M[(tt*N+n)][i*R+k]; fp32 state, bf16 M.
__global__ __launch_bounds__(128) void cwfa_phase2(const u16b* __restrict__ wsM,
                                                   float* __restrict__ h,
                                                   int Tc) {
  __shared__ float hs[128];
  const int n = blockIdx.x;
  const int k = threadIdx.x;
  hs[k] = h[n * 128 + k];
  __syncthreads();
  for (int tt = 0; tt < Tc; ++tt) {
    const u16b* Mp = wsM + (size_t)(tt * Ndim + n) * RR;
    float acc = 0.f;
    #pragma unroll 8
    for (int i = 0; i < 128; ++i)
      acc = fmaf(hs[i], bf2f(Mp[i * 128 + k]), acc);  // coalesced along k
    __syncthreads();
    hs[k] = acc;
    __syncthreads();
  }
  h[n * 128 + k] = hs[k];
}

// --- out[n][m] = sum_i h[n][i] * Omega[i][m]
__global__ __launch_bounds__(32) void cwfa_out(const float* __restrict__ h,
                                               const float* __restrict__ Om,
                                               float* __restrict__ out) {
  const int n = blockIdx.x;
  const int m = threadIdx.x;
  float acc = 0.f;
  #pragma unroll 8
  for (int i = 0; i < 128; ++i)
    acc = fmaf(h[n * 128 + i], Om[i * Mdim + m], acc);
  out[n * Mdim + m] = acc;
}

extern "C" void kernel_launch(void* const* d_in, const int* in_sizes, int n_in,
                              void* d_out, int out_size, void* d_ws, size_t ws_size,
                              hipStream_t stream) {
  const float* x     = (const float*)d_in[0];   // [N, L, D]
  const float* alpha = (const float*)d_in[1];   // [R]
  const float* A     = (const float*)d_in[2];   // [R, D, R]
  const float* Om    = (const float*)d_in[3];   // [R, M]
  float* out = (float*)d_out;                   // [N, M]

  char* ws = (char*)d_ws;
  u16b*  wsA = (u16b*)ws;                                   // 4 MB bf16 A_r
  float* h   = (float*)(ws + (size_t)(4 << 20));            // 64 KB fp32 state
  u16b*  wsM = (u16b*)(ws + (size_t)(4 << 20) + (64 << 10));
  const size_t baseBytes = (size_t)(4 << 20) + (64 << 10);
  const size_t perStep   = (size_t)Ndim * RR * sizeof(u16b);  // 4 MB / step

  // Largest power-of-two chunk whose M buffer fits the workspace (<=32 so the
  // chunk stays comfortably inside the 192 MB L2 as well).
  int Tc = 1;
  for (int c = 32; c >= 2; c >>= 1) {
    if (ws_size >= baseBytes + (size_t)c * perStep) { Tc = c; break; }
  }

  cwfa_convA <<<(Rdim * Ddim * Rdim) / 256, 256, 0, stream>>>(A, wsA);
  cwfa_init_h<<<(Ndim * Rdim) / 256,        256, 0, stream>>>(alpha, h);

  for (int t0 = 0; t0 < Ldim; t0 += Tc) {
    cwfa_phase1<<<dim3(RR / 128, Tc), 256, 0, stream>>>(x, wsA, wsM, t0);
    cwfa_phase2<<<Ndim, 128, 0, stream>>>(wsM, h, Tc);
  }
  cwfa_out<<<Ndim, Mdim, 0, stream>>>(h, Om, out);
}
